// Injector_807453852353
// MI455X (gfx1250) — compile-verified, matrix operand order checked by reading, not run on hardware
//
#include <hip/hip_runtime.h>
#include <math.h>

typedef __attribute__((ext_vector_type(16))) __bf16 v16bf_t;
typedef __attribute__((ext_vector_type(8)))  __bf16 v8bf_t;
typedef __attribute__((ext_vector_type(8)))  float  v8f_t;
typedef __attribute__((ext_vector_type(4)))  unsigned int u32x4_t;
typedef __attribute__((ext_vector_type(8)))  int i32x8_t;
typedef __attribute__((ext_vector_type(4)))  int i32x4_t;

namespace {
constexpr int kB  = 4;
constexpr int kLQ = 4096;
constexpr int kC  = 384;            // K for every GEMM
constexpr int kNH = 6;
constexpr int kNP = 4;
constexpr int kHL = 64;
constexpr int kWL = 64;
constexpr int kLV = kHL * kWL;      // 4096
constexpr int kDH = kC / kNH;       // 64
constexpr int kM  = kB * kLQ;       // 16384 rows
constexpr float kEps = 1e-6f;
}

// ---------------------------------------------------------------------------
// LayerNorm over C=384 -> bf16. 128 threads/row.
// ---------------------------------------------------------------------------
__global__ void mda_ln_bf16(const float* __restrict__ x,
                            const float* __restrict__ g,
                            const float* __restrict__ b,
                            __bf16* __restrict__ y, int rows) {
  int row = blockIdx.x;
  if (row >= rows) return;
  const float* xr = x + (size_t)row * kC;
  int t = threadIdx.x;
  float v0 = xr[t], v1 = xr[t + 128], v2 = xr[t + 256];

  __shared__ float red[128];
  red[t] = v0 + v1 + v2;
  __syncthreads();
  for (int off = 64; off > 0; off >>= 1) {
    if (t < off) red[t] += red[t + off];
    __syncthreads();
  }
  float mu = red[0] * (1.0f / kC);
  __syncthreads();

  float d0 = v0 - mu, d1 = v1 - mu, d2 = v2 - mu;
  red[t] = d0 * d0 + d1 * d1 + d2 * d2;
  __syncthreads();
  for (int off = 64; off > 0; off >>= 1) {
    if (t < off) red[t] += red[t + off];
    __syncthreads();
  }
  float rstd = rsqrtf(red[0] * (1.0f / kC) + kEps);

  __bf16* yr = y + (size_t)row * kC;
  yr[t]       = (__bf16)(d0 * rstd * g[t]       + b[t]);
  yr[t + 128] = (__bf16)(d1 * rstd * g[t + 128] + b[t + 128]);
  yr[t + 256] = (__bf16)(d2 * rstd * g[t + 256] + b[t + 256]);
}

// ---------------------------------------------------------------------------
// Weight prep: W (K x N f32 row-major) -> WT (Npad x K bf16), pad rows zeroed.
// ---------------------------------------------------------------------------
__global__ void mda_wT_bf16(const float* __restrict__ W, __bf16* __restrict__ WT,
                            int K, int N, int Npad) {
  int idx = blockIdx.x * blockDim.x + threadIdx.x;
  if (idx >= Npad * K) return;
  int n = idx / K, k = idx % K;
  float v = (n < N) ? W[(size_t)k * N + n] : 0.0f;
  WT[(size_t)n * K + k] = (__bf16)v;
}

// ---------------------------------------------------------------------------
// TDM helper: issue tensor_load_to_lds of a [rows x kC] bf16 strip into LDS.
// D# built per CDNA5 ISA (count=1, type=2, data_size=2B).
// ---------------------------------------------------------------------------
#if __has_builtin(__builtin_amdgcn_tensor_load_to_lds)
#define MDA_HAVE_TDM 1
__device__ __forceinline__ void mda_tdm_load_strip(const __bf16* gsrc,
                                                   void* lds_dst,
                                                   int rows, int totalRows) {
  unsigned long long ga = (unsigned long long)(uintptr_t)gsrc;
  unsigned lds = (unsigned)(uintptr_t)lds_dst;     // LDS aperture: low 32 bits

  u32x4_t g0;
  g0[0] = 1u;                                      // count=1, is_restore=0
  g0[1] = lds;                                     // lds_addr [63:32]
  g0[2] = (unsigned)(ga & 0xFFFFFFFFull);          // global_addr[31:0]
  g0[3] = (unsigned)((ga >> 32) & 0x01FFFFFFull)   // global_addr[56:32]
        | (2u << 30);                              // type=2 ("image")

  i32x8_t g1;
  g1[0] = (int)(1u << 16);                         // data_size=1 -> 2 bytes
  g1[1] = (int)(((unsigned)kC & 0xFFFFu) << 16);   // tensor_dim0[15:0]=384
  g1[2] = (int)(((unsigned)totalRows & 0xFFFFu) << 16); // tensor_dim1[15:0]
  g1[3] = (int)(((unsigned)kC & 0xFFFFu) << 16);   // tile_dim0=384
  g1[4] = rows;                                    // tile_dim1=rows
  g1[5] = kC;                                      // tensor_dim0_stride=384
  g1[6] = 0;
  g1[7] = 0;

  i32x4_t gz = {0, 0, 0, 0};
#if defined(__clang_major__) && (__clang_major__ >= 23)
  i32x8_t gz8 = {0, 0, 0, 0, 0, 0, 0, 0};
  __builtin_amdgcn_tensor_load_to_lds(g0, g1, gz, gz, gz8, 0);
#else
  __builtin_amdgcn_tensor_load_to_lds(g0, g1, gz, gz, 0);
#endif
  __builtin_amdgcn_s_wait_tensorcnt(0);
}
#else
#define MDA_HAVE_TDM 0
#endif

// ---------------------------------------------------------------------------
// WMMA GEMM: D[kM,Npad] = A[kM,384](bf16) x BT[Npad,384](bf16)^T (+bias).
// Block = 8 waves = 8 M-tiles sharing one N-strip of WN tiles. The B-strip
// (WN*16 x 384 bf16) is DMA'd to LDS once per block by the Tensor Data Mover,
// then each wave runs WN accumulators: per K-step, 2 global A loads + 2*WN
// ds_load_b128 feed WN v_wmma_f32_16x16x32_bf16. Grid is exact -> EXEC all-1s.
// If resid != nullptr: outF[m,n] = resid[m,n] + gamma[n] * (acc + bias[n]).
// ---------------------------------------------------------------------------
template <int WN>
__global__ void mda_gemm_wmma(const __bf16* __restrict__ A,
                              const __bf16* __restrict__ BT,
                              const float* __restrict__ bias,
                              float* __restrict__ D,
                              const float* __restrict__ resid,
                              const float* __restrict__ gamma,
                              float* __restrict__ outF,
                              int Npad, int Nvalid) {
  __shared__ __bf16 sB[WN * 16 * kC];              // up to 48 KB (WN=4)

  int wave = threadIdx.x >> 5;
  int lane = threadIdx.x & 31;
  int nStrips = (Npad >> 4) / WN;
  int mblk = blockIdx.x / nStrips;
  int sidx = blockIdx.x - mblk * nStrips;
  int m0 = (mblk * 8 + wave) << 4;
  int n0 = sidx * (WN * 16);

  const __bf16* gB = BT + (size_t)n0 * kC;
#if MDA_HAVE_TDM
  if (wave == 0)
    mda_tdm_load_strip(gB, sB, WN * 16, Npad);     // TDM -> s_wait_tensorcnt
#else
  {
    const uint4* src = (const uint4*)gB;
    uint4* dst = (uint4*)sB;
    int chunks = WN * 16 * kC / 8;                 // 8 bf16 per 16B chunk
    for (int i = threadIdx.x; i < chunks; i += 256) dst[i] = src[i];
  }
#endif
  __syncthreads();

  int half = lane >> 4;                            // 0: lanes 0-15, 1: 16-31
  int l16  = lane & 15;
  const __bf16* Arow = A + (size_t)(m0 + l16) * kC;

  v8f_t acc[WN];
#pragma unroll
  for (int j = 0; j < WN; ++j) acc[j] = v8f_t{};

  for (int k0 = 0; k0 < kC; k0 += 32) {
    __builtin_prefetch(Arow + k0 + 128, 0, 0);     // global_prefetch_b8
    v8bf_t a_lo = *(const v8bf_t*)(Arow + k0 + half * 8);
    v8bf_t a_hi = *(const v8bf_t*)(Arow + k0 + 16 + half * 8);
    v16bf_t av;
#pragma unroll
    for (int i = 0; i < 8; ++i) { av[i] = a_lo[i]; av[i + 8] = a_hi[i]; }

#pragma unroll
    for (int j = 0; j < WN; ++j) {
      const __bf16* Brow = sB + (size_t)(j * 16 + l16) * kC;
      v8bf_t b_lo = *(const v8bf_t*)(Brow + k0 + half * 8);     // ds_load_b128
      v8bf_t b_hi = *(const v8bf_t*)(Brow + k0 + 16 + half * 8);
      v16bf_t bv;
#pragma unroll
      for (int i = 0; i < 8; ++i) { bv[i] = b_lo[i]; bv[i + 8] = b_hi[i]; }
      acc[j] = __builtin_amdgcn_wmma_f32_16x16x32_bf16(
          false, av, false, bv, (short)0, acc[j], false, false);
    }
  }

#pragma unroll
  for (int j = 0; j < WN; ++j) {
    int n = n0 + j * 16 + l16;
    float bv_ = (bias != nullptr && n < Nvalid) ? bias[n] : 0.0f;
    float gv  = (gamma != nullptr && n < Nvalid) ? gamma[n] : 0.0f;
#pragma unroll
    for (int r = 0; r < 8; ++r) {
      int m = m0 + (half << 3) + r;                // C/D layout: VGPR r -> M
      float v = acc[j][r] + bv_;
      if (resid != nullptr) {
        if (n < Nvalid)
          outF[(size_t)m * Nvalid + n] = resid[(size_t)m * Nvalid + n] + gv * v;
      } else {
        D[(size_t)m * Npad + n] = v;
      }
    }
  }
}

// ---------------------------------------------------------------------------
// Deformable sampling: one wave per (b, q, h). Softmax over 4 points,
// bilinear gather from value (25 MB, L2-resident on a 192 MB L2), 2 floats
// per lane (DH=64). Output bf16 row feeding the final WMMA GEMM.
// ---------------------------------------------------------------------------
__global__ void mda_sample(const float* __restrict__ value,    // (b*LV+pix)*C + h*DH + d
                           const float* __restrict__ offsets,  // (B*LQ, 48)
                           const float* __restrict__ logits,   // (B*LQ, 32) padded
                           const float* __restrict__ refpts,   // (B*LQ, 2)  (NL==1)
                           __bf16* __restrict__ mid) {         // (B*LQ, C)
  int wave = threadIdx.x >> 5;
  int lane = threadIdx.x & 31;
  int widx = blockIdx.x * (blockDim.x >> 5) + wave;            // over B*LQ*NH
  if (widx >= kM * kNH) return;

  int h = widx % kNH;
  int row = widx / kNH;                                        // b*LQ + q
  int b = row / kLQ;

  const float* off = offsets + (size_t)row * 48 + h * 8;
  const float* lg  = logits  + (size_t)row * 32 + h * 4;

  float w0 = lg[0], w1 = lg[1], w2 = lg[2], w3 = lg[3];
  float mx = fmaxf(fmaxf(w0, w1), fmaxf(w2, w3));
  float e0 = __expf(w0 - mx), e1 = __expf(w1 - mx);
  float e2 = __expf(w2 - mx), e3 = __expf(w3 - mx);
  float inv = 1.0f / (e0 + e1 + e2 + e3);
  float aw[4] = {e0 * inv, e1 * inv, e2 * inv, e3 * inv};

  float rx = refpts[(size_t)row * 2 + 0];
  float ry = refpts[(size_t)row * 2 + 1];

  int d = lane * 2;
  const float* vb = value + (size_t)b * kLV * kC + h * kDH + d;

  float acc0 = 0.0f, acc1 = 0.0f;
#pragma unroll
  for (int p = 0; p < kNP; ++p) {
    float x = rx * kWL + off[p * 2 + 0] - 0.5f;   // (rx + ox/W)*W - 0.5
    float y = ry * kHL + off[p * 2 + 1] - 0.5f;
    float xf = floorf(x), yf = floorf(y);
    int x0 = (int)xf, y0 = (int)yf;
    float wx = x - xf, wy = y - yf;
    float cw[4] = {(1.0f - wx) * (1.0f - wy), wx * (1.0f - wy),
                   (1.0f - wx) * wy,          wx * wy};
    int cx[4] = {x0, x0 + 1, x0,     x0 + 1};
    int cy[4] = {y0, y0,     y0 + 1, y0 + 1};
#pragma unroll
    for (int c = 0; c < 4; ++c) {
      int xi = cx[c], yi = cy[c];
      if (xi >= 0 && xi < kWL && yi >= 0 && yi < kHL) {        // wave-uniform
        const float* vp = vb + (size_t)(yi * kWL + xi) * kC;
        float s = aw[p] * cw[c];
        acc0 += s * vp[0];
        acc1 += s * vp[1];
      }
    }
  }
  __bf16* mr = mid + (size_t)row * kC + h * kDH + d;
  mr[0] = (__bf16)acc0;
  mr[1] = (__bf16)acc1;
}

// ---------------------------------------------------------------------------
// Host launcher
// ---------------------------------------------------------------------------
extern "C" void kernel_launch(void* const* d_in, const int* in_sizes, int n_in,
                              void* d_out, int out_size, void* d_ws, size_t ws_size,
                              hipStream_t stream) {
  const float* query = (const float*)d_in[0];
  const float* refp  = (const float*)d_in[1];
  const float* feat  = (const float*)d_in[2];
  // d_in[3] spatial_shapes, d_in[4] level_start_index: compile-time constants
  const float* ln_q_g = (const float*)d_in[5];
  const float* ln_q_b = (const float*)d_in[6];
  const float* ln_f_g = (const float*)d_in[7];
  const float* ln_f_b = (const float*)d_in[8];
  const float* gamma  = (const float*)d_in[9];
  const float* Wv     = (const float*)d_in[10];
  const float* bvv    = (const float*)d_in[11];
  const float* Wo     = (const float*)d_in[12];
  const float* bo     = (const float*)d_in[13];
  const float* Wa     = (const float*)d_in[14];
  const float* ba     = (const float*)d_in[15];
  const float* Wout   = (const float*)d_in[16];
  const float* bout   = (const float*)d_in[17];
  float* out = (float*)d_out;

  // Workspace carve-up (256B aligned chunks)
  char* w = (char*)d_ws;
  size_t off = 0;
  auto alloc = [&](size_t bytes) -> void* {
    void* p = w + off;
    off += (bytes + 255) & ~(size_t)255;
    return p;
  };
  __bf16* qln   = (__bf16*)alloc((size_t)kM * kC * 2);
  __bf16* fln   = (__bf16*)alloc((size_t)kM * kC * 2);
  float*  value = (float*) alloc((size_t)kM * kC * 4);
  float*  offs  = (float*) alloc((size_t)kM * 48 * 4);
  float*  logit = (float*) alloc((size_t)kM * 32 * 4);
  __bf16* mid   = (__bf16*)alloc((size_t)kM * kC * 2);
  __bf16* WvT   = (__bf16*)alloc((size_t)kC * kC * 2);
  __bf16* WoutT = (__bf16*)alloc((size_t)kC * kC * 2);
  __bf16* WoT   = (__bf16*)alloc((size_t)48 * kC * 2);
  __bf16* WaT   = (__bf16*)alloc((size_t)32 * kC * 2);

  auto cdiv = [](int a, int b) { return (a + b - 1) / b; };

  // 1) weight transpose + bf16 convert (Wa cols padded 24 -> 32 with zeros)
  mda_wT_bf16<<<cdiv(384 * kC, 256), 256, 0, stream>>>(Wv,   WvT,   kC, kC, 384);
  mda_wT_bf16<<<cdiv(384 * kC, 256), 256, 0, stream>>>(Wout, WoutT, kC, kC, 384);
  mda_wT_bf16<<<cdiv(48  * kC, 256), 256, 0, stream>>>(Wo,   WoT,   kC, 48, 48);
  mda_wT_bf16<<<cdiv(32  * kC, 256), 256, 0, stream>>>(Wa,   WaT,   kC, 24, 32);

  // 2) LayerNorms -> bf16 activations
  mda_ln_bf16<<<kM, 128, 0, stream>>>(query, ln_q_g, ln_q_b, qln, kM);
  mda_ln_bf16<<<kM, 128, 0, stream>>>(feat,  ln_f_g, ln_f_b, fln, kM);

  const int mBlocks = kM / 16 / 8;                 // 128

  // 3) value = f_ln @ Wv + bv   (N=384: 24 tiles, WN=4 -> 6 strips)
  mda_gemm_wmma<4><<<mBlocks * 6, 256, 0, stream>>>(
      fln, WvT, bvv, value, nullptr, nullptr, nullptr, 384, 384);

  // 4) offsets = q_ln @ Wo + bo  (N=48: 3 tiles, WN=3 -> 1 strip)
  mda_gemm_wmma<3><<<mBlocks * 1, 256, 0, stream>>>(
      qln, WoT, bo, offs, nullptr, nullptr, nullptr, 48, 48);

  // 5) attn logits = q_ln @ Wa + ba  (N=24 padded to 32: 2 tiles, WN=2)
  mda_gemm_wmma<2><<<mBlocks * 1, 256, 0, stream>>>(
      qln, WaT, ba, logit, nullptr, nullptr, nullptr, 32, 24);

  // 6) softmax + bilinear sampling -> mid (bf16)
  mda_sample<<<cdiv(kM * kNH, 8), 256, 0, stream>>>(value, offs, logit, refp, mid);

  // 7) out = query + gamma * (mid @ Wout + bout)   (fused epilogue)
  mda_gemm_wmma<4><<<mBlocks * 6, 256, 0, stream>>>(
      mid, WoutT, bout, nullptr, query, gamma, out, 384, 384);
}
